// Attention_12060268167989
// MI455X (gfx1250) — compile-verified, matrix operand order checked by reading, not run on hardware
//
#include <hip/hip_runtime.h>

// Problem constants (match reference)
#define T_SEQ   4096
#define D_MODEL 2048
#define N_HEADS 16
#define KV_HEADS 4
#define HEAD_DIM 128
#define WINDOW_SZ 512
#define HHD (N_HEADS * HEAD_DIM)    // 2048
#define KVD (KV_HEADS * HEAD_DIM)   // 512
#define N_REP (N_HEADS / KV_HEADS)  // 4

#define SCALE_F 0.08838834764831845f   // 1/sqrt(128)
#define LOG2E   1.4426950408889634f
#define NEG_BIG (-1e30f)

typedef __attribute__((ext_vector_type(16))) __bf16 v16bf;
typedef __attribute__((ext_vector_type(8)))  __bf16 v8bf;
typedef __attribute__((ext_vector_type(8)))  float  v8f;

// ---------------------------------------------------------------------------
// WMMA helpers (CDNA5: v_wmma_f32_16x16x32_bf16, D = A(16x32) * B(32x16) + C)
// ---------------------------------------------------------------------------
static __device__ __forceinline__ v8f wmma_bf16f32(v16bf a, v16bf b, v8f c) {
  return __builtin_amdgcn_wmma_f32_16x16x32_bf16(
      /*neg_a=*/false, a, /*neg_b=*/false, b,
      /*c_mod=*/(short)0, c, /*reuse_a=*/false, /*reuse_b=*/false);
}

// A-fragment address for a 16x32 bf16 tile of a row-major matrix.
// Lane L: row = row0 + (L&15); cols = col0 + (L>>4)*8 + {0..7} and +16+{0..7}.
static __device__ __forceinline__ const __bf16*
frag_ptr(const __bf16* base, long row0, long col0, int lane, long ld) {
  long r = row0 + (lane & 15);
  long c = col0 + (long)((lane >> 4) << 3);
  return base + r * ld + c;
}

static __device__ __forceinline__ v16bf load_frag(const __bf16* p) {
  v8bf lo = *(const v8bf*)(p);        // cols c..c+7   (global_load_b128)
  v8bf hi = *(const v8bf*)(p + 16);   // cols c+16..c+23
  v16bf f;
#pragma unroll
  for (int i = 0; i < 8; ++i) { f[i] = lo[i]; f[i + 8] = hi[i]; }
  return f;
}

// ---------------------------------------------------------------------------
// Conversion kernels: fp32 -> bf16, optionally transposed (for B operands)
// ---------------------------------------------------------------------------
__global__ void cvt_bf16_kernel(const float* __restrict__ in,
                                __bf16* __restrict__ out, long n) {
  long i = (long)blockIdx.x * blockDim.x + threadIdx.x;
  long stride = (long)gridDim.x * blockDim.x;
  for (; i < n; i += stride) out[i] = (__bf16)in[i];
}

// in: R x C row-major (fp32)  ->  out: C x R row-major (bf16)
__global__ void transpose_cvt_kernel(const float* __restrict__ in,
                                     __bf16* __restrict__ out, int R, int C) {
  long n = (long)R * C;
  long i = (long)blockIdx.x * blockDim.x + threadIdx.x;
  long stride = (long)gridDim.x * blockDim.x;
  for (; i < n; i += stride) {
    long rr = i % R;   // column of out == row of in
    long cc = i / R;   // row of out == column of in
    out[i] = (__bf16)in[rr * C + cc];
  }
}

// ---------------------------------------------------------------------------
// Generic WMMA GEMM: C[M,N] = A[M,K] (bf16, row-major) * Bt[N,K]^T (bf16)
// 256 threads = 8 waves arranged 4(M) x 2(N); each wave computes a 64x64 tile
// (4x4 wmma sub-tiles): 16 wmma per K=32 step against 8 fragment loads.
// Block tile: 256 (M) x 128 (N). mode: 0 = bf16 row-major store,
// 1 = bf16 transposed store (out[n*M+m], packed b128), 2 = f32 row-major.
// ---------------------------------------------------------------------------
__global__ void __launch_bounds__(256)
gemm_bf16_kernel(const __bf16* __restrict__ A, const __bf16* __restrict__ Bt,
                 void* __restrict__ Cout, int M, int N, int K, int mode) {
  const int lane = threadIdx.x & 31;
  const int wave = threadIdx.x >> 5;
  const int wm = wave >> 1;                 // 0..3
  const int wn = wave & 1;                  // 0..1
  const long row0 = (long)blockIdx.y * 256 + (long)wm * 64;
  const long col0 = (long)blockIdx.x * 128 + (long)wn * 64;

  v8f c[4][4];
#pragma unroll
  for (int i = 0; i < 4; ++i)
#pragma unroll
    for (int j = 0; j < 4; ++j)
#pragma unroll
      for (int r = 0; r < 8; ++r) c[i][j][r] = 0.0f;

  for (int kk = 0; kk < K; kk += 32) {
    v16bf a[4], b[4];
#pragma unroll
    for (int i = 0; i < 4; ++i) {
      const __bf16* pa = frag_ptr(A, row0 + (long)i * 16, kk, lane, K);
      __builtin_prefetch(pa + 32, 0, 1);    // global_prefetch_b8, next K step
      a[i] = load_frag(pa);
    }
#pragma unroll
    for (int j = 0; j < 4; ++j) {
      const __bf16* pb = frag_ptr(Bt, col0 + (long)j * 16, kk, lane, K);
      __builtin_prefetch(pb + 32, 0, 1);
      b[j] = load_frag(pb);
    }
#pragma unroll
    for (int i = 0; i < 4; ++i)
#pragma unroll
      for (int j = 0; j < 4; ++j)
        c[i][j] = wmma_bf16f32(a[i], b[j], c[i][j]);
  }

  // C/D layout: lane holds column n = (lane&15), rows m = r + 8*(lane>>4)
  const int nlane = lane & 15;
  const long rbase = (long)((lane >> 4) << 3);
#pragma unroll
  for (int i = 0; i < 4; ++i) {
#pragma unroll
    for (int j = 0; j < 4; ++j) {
      long m0 = row0 + (long)i * 16 + rbase;
      long nn = col0 + (long)j * 16 + nlane;
      if (mode == 0) {
        __bf16* Cb = (__bf16*)Cout;
#pragma unroll
        for (int r = 0; r < 8; ++r)
          Cb[(m0 + r) * (long)N + nn] = (__bf16)c[i][j][r];
      } else if (mode == 1) {
        __bf16* Cb = (__bf16*)Cout;          // store transposed: out[n*M + m]
        v8bf pk;
#pragma unroll
        for (int r = 0; r < 8; ++r) pk[r] = (__bf16)c[i][j][r];
        *(v8bf*)(Cb + nn * (long)M + m0) = pk;   // m0 % 8 == 0 -> 16B aligned
      } else {
        float* Cf = (float*)Cout;
#pragma unroll
        for (int r = 0; r < 8; ++r)
          Cf[(m0 + r) * (long)N + nn] = c[i][j][r];
      }
    }
  }
}

// ---------------------------------------------------------------------------
// Flash attention with sliding window (GQA).
// One wave per (head, 32-query tile) = two 16-row query sub-tiles, so each
// K fragment feeds 2 S-wmmas and each V fragment feeds 2 O-wmmas (1:1
// wmma-to-b128 ratio). Computes S^T = K*Q^T so the f32 accumulator layout
// (lane = query, regs = keys) maps in-lane to the bf16 P fragment, and
// O^T = V^T * P^T so softmax rescale/normalize is a per-lane scalar op.
// Qb: [T, HHD] bf16 ; Kb: [T, KVD] bf16 ; Vt: [KVD, T] bf16 (transposed)
// attn out: [T, HHD] bf16.
// ---------------------------------------------------------------------------
__global__ void __launch_bounds__(128)
attn_kernel(const __bf16* __restrict__ Qb, const __bf16* __restrict__ Kb,
            const __bf16* __restrict__ Vt, __bf16* __restrict__ attn) {
  const int lane = threadIdx.x & 31;
  const int wave = threadIdx.x >> 5;
  const int wid  = blockIdx.x * 4 + wave;          // 0 .. H*(T/32)-1
  const int wt   = wid & (T_SEQ / 32 - 1);         // 32-query tile (128 tiles)
  const int h    = wid >> 7;                       // head
  const int kvh  = h / N_REP;
  const int q0   = wt * 32;
  const int half8 = (lane >> 4) << 3;
  const int qg0  = q0 + (lane & 15);               // query of sub-tile 0
  const int qg1  = qg0 + 16;                       // query of sub-tile 1

  // Preload Q fragments for both query sub-tiles.
  v16bf qf[2][4];
#pragma unroll
  for (int tq = 0; tq < 2; ++tq)
#pragma unroll
    for (int cch = 0; cch < 4; ++cch)
      qf[tq][cch] = load_frag(
          frag_ptr(Qb, q0 + tq * 16, (long)h * HEAD_DIM + cch * 32, lane, HHD));

  // O^T accumulators: per sub-tile, 8 tiles of 16(hd) x 16(q).
  v8f o[2][8];
#pragma unroll
  for (int tq = 0; tq < 2; ++tq)
#pragma unroll
    for (int t = 0; t < 8; ++t)
#pragma unroll
      for (int r = 0; r < 8; ++r) o[tq][t][r] = 0.0f;

  float m0 = NEG_BIG, l0 = 0.0f, m1 = NEG_BIG, l1 = 0.0f;

  const int kb0 = (q0 >= WINDOW_SZ) ? ((q0 - WINDOW_SZ) & ~31) : 0;
  const __bf16* Vh = Vt + (long)kvh * HEAD_DIM * T_SEQ;

  for (int kb = kb0; kb <= q0 + 31; kb += 32) {
    // --- S^T = K_tile * Q^T : 4 C frags (2 query sub-tiles x 2 key halves) ---
    v8f s[2][2];
#pragma unroll
    for (int tq = 0; tq < 2; ++tq)
#pragma unroll
      for (int sk = 0; sk < 2; ++sk)
#pragma unroll
        for (int r = 0; r < 8; ++r) s[tq][sk][r] = 0.0f;
#pragma unroll
    for (int cch = 0; cch < 4; ++cch) {
      v16bf k0 = load_frag(frag_ptr(Kb, kb,      (long)kvh * HEAD_DIM + cch * 32, lane, KVD));
      v16bf k1 = load_frag(frag_ptr(Kb, kb + 16, (long)kvh * HEAD_DIM + cch * 32, lane, KVD));
      s[0][0] = wmma_bf16f32(k0, qf[0][cch], s[0][0]);
      s[0][1] = wmma_bf16f32(k1, qf[0][cch], s[0][1]);
      s[1][0] = wmma_bf16f32(k0, qf[1][cch], s[1][0]);
      s[1][1] = wmma_bf16f32(k1, qf[1][cch], s[1][1]);
    }

    // --- per sub-tile: scale + window mask + online softmax -> P fragment ---
    v16bf pf[2];
#pragma unroll
    for (int tq = 0; tq < 2; ++tq) {
      const int qg = tq ? qg1 : qg0;
      float mm = tq ? m1 : m0;
      float ll = tq ? l1 : l0;

      float vmax = NEG_BIG;
#pragma unroll
      for (int r = 0; r < 8; ++r) {
        int k0i = kb + half8 + r;
        int k1i = k0i + 16;
        float a0 = s[tq][0][r] * SCALE_F;
        float a1 = s[tq][1][r] * SCALE_F;
        a0 = ((k0i <= qg) && (qg - k0i <= WINDOW_SZ)) ? a0 : NEG_BIG;
        a1 = ((k1i <= qg) && (qg - k1i <= WINDOW_SZ)) ? a1 : NEG_BIG;
        s[tq][0][r] = a0; s[tq][1][r] = a1;
        vmax = fmaxf(vmax, fmaxf(a0, a1));
      }
      // lanes L and L+16 hold complementary key subsets for the same query
      vmax = fmaxf(vmax, __shfl_xor(vmax, 16, 32));
      float mnew  = fmaxf(mm, vmax);
      float alpha = __builtin_exp2f((mm - mnew) * LOG2E);

      float lsum = 0.0f;
#pragma unroll
      for (int r = 0; r < 8; ++r) {
        float p0 = __builtin_exp2f((s[tq][0][r] - mnew) * LOG2E);
        float p1 = __builtin_exp2f((s[tq][1][r] - mnew) * LOG2E);
        lsum += p0 + p1;
        pf[tq][r]     = (__bf16)p0;   // keys kb + half8 + r
        pf[tq][r + 8] = (__bf16)p1;   // keys kb + 16 + half8 + r
      }
      lsum += __shfl_xor(lsum, 16, 32);
      ll = ll * alpha + lsum;
      if (tq) { m1 = mnew; l1 = ll; } else { m0 = mnew; l0 = ll; }

      // rescale running O^T (per-lane scalar, thanks to O^T layout)
#pragma unroll
      for (int t = 0; t < 8; ++t)
#pragma unroll
        for (int r = 0; r < 8; ++r) o[tq][t][r] *= alpha;
    }

    // --- O^T += V^T(hd x 32keys) * P^T : V fragment reused by both tiles ---
#pragma unroll
    for (int t = 0; t < 8; ++t) {
      v16bf vf = load_frag(frag_ptr(Vh, (long)t * 16, kb, lane, T_SEQ));
      o[0][t] = wmma_bf16f32(vf, pf[0], o[0][t]);
      o[1][t] = wmma_bf16f32(vf, pf[1], o[1][t]);
    }
  }

  // --- normalize and store: lane holds one query, 8 consecutive hd per tile ---
#pragma unroll
  for (int tq = 0; tq < 2; ++tq) {
    float inv = 1.0f / (tq ? l1 : l0);
    const int qg = tq ? qg1 : qg0;
    __bf16* orow = attn + (long)qg * HHD + (long)h * HEAD_DIM;
#pragma unroll
    for (int t = 0; t < 8; ++t) {
      v8bf ob;
#pragma unroll
      for (int r = 0; r < 8; ++r) ob[r] = (__bf16)(o[tq][t][r] * inv);
      *(v8bf*)(orow + t * 16 + half8) = ob;   // 16B-aligned packed store
    }
  }
}

// ---------------------------------------------------------------------------
// Host-side orchestration (graph-capture safe: launches only)
// ---------------------------------------------------------------------------
extern "C" void kernel_launch(void* const* d_in, const int* in_sizes, int n_in,
                              void* d_out, int out_size, void* d_ws, size_t ws_size,
                              hipStream_t stream) {
  (void)in_sizes; (void)n_in; (void)out_size; (void)ws_size;
  const float* x  = (const float*)d_in[0];   // [T, D]
  const float* Wq = (const float*)d_in[1];   // [D, HHD]
  const float* Wk = (const float*)d_in[2];   // [D, KVD]
  const float* Wv = (const float*)d_in[3];   // [D, KVD]
  const float* Wo = (const float*)d_in[4];   // [HHD, D]
  float* out = (float*)d_out;                // [T, D]

  unsigned char* ws = (unsigned char*)d_ws;
  size_t off = 0;
  auto carve = [&](size_t bytes) -> void* {
    void* p = (void*)(ws + off);
    off += (bytes + 255) & ~(size_t)255;
    return p;
  };
  __bf16* xb   = (__bf16*)carve((size_t)T_SEQ * D_MODEL * 2);  // x bf16
  __bf16* WqT  = (__bf16*)carve((size_t)HHD * D_MODEL * 2);    // Wq^T bf16
  __bf16* WkT  = (__bf16*)carve((size_t)KVD * D_MODEL * 2);
  __bf16* WvT  = (__bf16*)carve((size_t)KVD * D_MODEL * 2);
  __bf16* WoT  = (__bf16*)carve((size_t)D_MODEL * HHD * 2);    // Wo^T bf16
  __bf16* Qb   = (__bf16*)carve((size_t)T_SEQ * HHD * 2);      // Q
  __bf16* Kb   = (__bf16*)carve((size_t)T_SEQ * KVD * 2);      // K
  __bf16* Vt   = (__bf16*)carve((size_t)KVD * T_SEQ * 2);      // V^T [kvh][hd][t]
  __bf16* attn = (__bf16*)carve((size_t)T_SEQ * HHD * 2);      // attention out

  // 1) precision conversion / weight transposition
  cvt_bf16_kernel<<<4096, 256, 0, stream>>>(x, xb, (long)T_SEQ * D_MODEL);
  transpose_cvt_kernel<<<4096, 256, 0, stream>>>(Wq, WqT, D_MODEL, HHD);
  transpose_cvt_kernel<<<2048, 256, 0, stream>>>(Wk, WkT, D_MODEL, KVD);
  transpose_cvt_kernel<<<2048, 256, 0, stream>>>(Wv, WvT, D_MODEL, KVD);
  transpose_cvt_kernel<<<4096, 256, 0, stream>>>(Wo, WoT, HHD, D_MODEL);

  // 2) projections (WMMA GEMMs). Block tile 256(M) x 128(N), 64x64 per wave.
  {
    dim3 g(HHD / 128, T_SEQ / 256);   // Q: 4096x2048x2048
    gemm_bf16_kernel<<<g, 256, 0, stream>>>(xb, WqT, Qb, T_SEQ, HHD, D_MODEL, 0);
  }
  {
    dim3 g(KVD / 128, T_SEQ / 256);   // K: 4096x512x2048
    gemm_bf16_kernel<<<g, 256, 0, stream>>>(xb, WkT, Kb, T_SEQ, KVD, D_MODEL, 0);
  }
  {
    dim3 g(KVD / 128, T_SEQ / 256);   // V: stored transposed -> Vt
    gemm_bf16_kernel<<<g, 256, 0, stream>>>(xb, WvT, Vt, T_SEQ, KVD, D_MODEL, 1);
  }

  // 3) sliding-window flash attention: 1 wave per (head, 32-query tile)
  {
    int nwaves = N_HEADS * (T_SEQ / 32);   // 2048 waves
    attn_kernel<<<nwaves / 4, 128, 0, stream>>>(Qb, Kb, Vt, attn);
  }

  // 4) output projection (f32 store into d_out)
  {
    dim3 g(D_MODEL / 128, T_SEQ / 256);
    gemm_bf16_kernel<<<g, 256, 0, stream>>>(attn, WoT, out, T_SEQ, D_MODEL, HHD, 2);
  }
}
